// FactorGNN_29712583754345
// MI455X (gfx1250) — compile-verified, matrix operand order checked by reading, not run on hardware
//
#include <hip/hip_runtime.h>
#include <hip/hip_bf16.h>

#define BN   4
#define NN   1024
#define FF   10
#define DD   16
#define DOUT 16
#define NF   10240   // N * F

typedef __attribute__((ext_vector_type(16))) __bf16        v16bf;
typedef __attribute__((ext_vector_type(8)))  float         v8f;
typedef __attribute__((ext_vector_type(8)))  unsigned int  v8u;
typedef __attribute__((ext_vector_type(4)))  unsigned int  v4u;

union ABFrag { v8u u; v16bf bf; };

__device__ __forceinline__ unsigned pack_bf16(float lo, float hi) {
    unsigned ulo = __builtin_bit_cast(unsigned, lo);
    unsigned uhi = __builtin_bit_cast(unsigned, hi);
    return (uhi & 0xFFFF0000u) | (ulo >> 16);
}

__device__ __forceinline__ unsigned short f32_to_bf16_rne(float x) {
    unsigned u = __builtin_bit_cast(unsigned, x);
    u += 0x7FFFu + ((u >> 16) & 1u);
    return (unsigned short)(u >> 16);
}

__device__ __forceinline__ float bf16_to_f32(unsigned short h) {
    return __builtin_bit_cast(float, (unsigned)h << 16);
}

__device__ __forceinline__ ABFrag load_bfrag(const unsigned short* p) {
    ABFrag f;
    v4u lo = *(const v4u*)(p);
    v4u hi = *(const v4u*)(p + 16);
    f.u = (v8u){lo.x, lo.y, lo.z, lo.w, hi.x, hi.y, hi.z, hi.w};
    return f;
}

// Raw f32 A-tile loads for one 16x32 fragment slice held by this lane.
// ra[0..7]  = rows (i0+8h) + 0..7   (K = 8h .. 8h+7)
// ra[8..15] = rows (i0+8h) + 16..23 (K = 16+8h .. 16+8h+7)
__device__ __forceinline__ void load_araw(const float* __restrict__ ap, float* ra) {
#pragma unroll
    for (int k = 0; k < 8; ++k) {
        ra[k]     = ap[(size_t)k * NF];
        ra[8 + k] = ap[(size_t)(16 + k) * NF];
    }
}

__device__ __forceinline__ ABFrag pack_afrag(const float* ra) {
    ABFrag a;
#pragma unroll
    for (int r = 0; r < 4; ++r) {
        a.u[r]     = pack_bf16(ra[2 * r],     ra[2 * r + 1]);
        a.u[4 + r] = pack_bf16(ra[8 + 2 * r], ra[8 + 2 * r + 1]);
    }
    return a;
}

// ---- Kernel A1: partial column sums of A (with self-loop fix) ----
__global__ void deg_partial(const float* __restrict__ S, float* __restrict__ deg) {
    int j  = blockIdx.x * 256 + threadIdx.x;
    int i0 = blockIdx.y * (NF / 32);
    float s = 0.f;
    for (int i = i0; i < i0 + NF / 32; ++i) {
        float a = S[(size_t)i * NF + j];
        if (i == j && a == 0.f) a = 1.f;
        s += a;
    }
    atomicAdd(&deg[j], s);
}

// ---- Kernel A2: deg -> d^{-1/2}, and diagonal-fix delta flag ----
__global__ void finalize_deg(const float* __restrict__ S, float* __restrict__ dinv,
                             float* __restrict__ dfix) {
    int j = blockIdx.x * 256 + threadIdx.x;
    if (j >= NF) return;
    float d = dinv[j];                       // holds deg on entry
    dinv[j] = d > 0.f ? rsqrtf(d) : 0.f;
    dfix[j] = (S[(size_t)j * NF + j] == 0.f) ? 1.f : 0.f;
}

// ---- Kernel B: HsT[c, i] = bf16( dinv[i] * (xf[b,i,:] @ W_gcn)[o] ), c = b*16+o ----
__global__ void gcn_linear(const float* __restrict__ x, const float* __restrict__ Wg,
                           const float* __restrict__ dinv, unsigned short* __restrict__ HsT) {
    int t = blockIdx.x * blockDim.x + threadIdx.x;
    if (t >= 64 * NF) return;
    int i = t % NF;
    int c = t / NF;
    int b = c >> 4;
    int o = c & 15;
    const float* xr = x + ((size_t)b * NF + i) * DD;
    float s = 0.f;
#pragma unroll
    for (int d = 0; d < DD; ++d) s += xr[d] * Wg[d * DOUT + o];
    HsT[(size_t)c * NF + i] = f32_to_bf16_rne(s * dinv[i]);
}

// ---- Kernel C: Out[j, c] = relu( dinv[j] * (sum_i A[i,j]*HsT[c,i]) + b_gcn[o] ) ----
// One 16-row j-tile per 4-wave workgroup; 4-way K split + LDS reduction.
// bf16 WMMA with f32 accumulate; A converted f32->bf16 in-register.
// Software double-buffer: iteration t+1's loads are issued before iteration t's WMMAs.
__global__ __launch_bounds__(128) void gcn_agg_wmma(
    const float* __restrict__ S, const unsigned short* __restrict__ HsT,
    const float* __restrict__ dinv, const float* __restrict__ dfix,
    const float* __restrict__ b_gcn, float* __restrict__ Hrelu) {
    __shared__ float red[4 * 1024];          // [wave][frag*256 + reg*32 + lane]
    const int lane = threadIdx.x & 31;
    const int wave = threadIdx.x >> 5;
    const int n    = lane & 15;              // M (A rows j) / N (B cols c) in-tile index
    const int half = lane >> 4;
    const int j0   = blockIdx.x * 16;

    const int KCH   = NF / 4;                // 2560 per wave
    const int ibase = wave * KCH;

    const float*          ap = S   + (size_t)(ibase + 8 * half) * NF + (j0 + n);
    const unsigned short* bp = HsT + (size_t)n * NF + ibase + 8 * half;
    const size_t          astep = (size_t)32 * NF;

    v8f acc0 = {}; v8f acc1 = {}; v8f acc2 = {}; v8f acc3 = {};

    // ---- prologue: tile 0 ----
    float ra[16];
    load_araw(ap, ra);
    ABFrag b0 = load_bfrag(bp + (size_t)0 * 16 * NF);
    ABFrag b1 = load_bfrag(bp + (size_t)1 * 16 * NF);
    ABFrag b2 = load_bfrag(bp + (size_t)2 * 16 * NF);
    ABFrag b3 = load_bfrag(bp + (size_t)3 * 16 * NF);
    ABFrag a  = pack_afrag(ra);
    ap += astep;
    bp += 32;

    for (int it = 32; it < KCH; it += 32) {
        __builtin_prefetch(ap + astep, 0, 3);         // 2 tiles ahead

        // issue next tile's loads first (retire behind current WMMAs)
        float rn[16];
        load_araw(ap, rn);
        ABFrag n0 = load_bfrag(bp + (size_t)0 * 16 * NF);
        ABFrag n1 = load_bfrag(bp + (size_t)1 * 16 * NF);
        ABFrag n2 = load_bfrag(bp + (size_t)2 * 16 * NF);
        ABFrag n3 = load_bfrag(bp + (size_t)3 * 16 * NF);

        // compute on current tile (operands loaded one full iteration ago)
        acc0 = __builtin_amdgcn_wmma_f32_16x16x32_bf16(false, a.bf, false, b0.bf, (short)0, acc0, false, false);
        acc1 = __builtin_amdgcn_wmma_f32_16x16x32_bf16(false, a.bf, false, b1.bf, (short)0, acc1, false, false);
        acc2 = __builtin_amdgcn_wmma_f32_16x16x32_bf16(false, a.bf, false, b2.bf, (short)0, acc2, false, false);
        acc3 = __builtin_amdgcn_wmma_f32_16x16x32_bf16(false, a.bf, false, b3.bf, (short)0, acc3, false, false);

        // rotate buffers; A pack for next tile happens after WMMAs were issued
        a = pack_afrag(rn);
        b0 = n0; b1 = n1; b2 = n2; b3 = n3;
        ap += astep;
        bp += 32;
    }

    // ---- epilogue tile ----
    acc0 = __builtin_amdgcn_wmma_f32_16x16x32_bf16(false, a.bf, false, b0.bf, (short)0, acc0, false, false);
    acc1 = __builtin_amdgcn_wmma_f32_16x16x32_bf16(false, a.bf, false, b1.bf, (short)0, acc1, false, false);
    acc2 = __builtin_amdgcn_wmma_f32_16x16x32_bf16(false, a.bf, false, b2.bf, (short)0, acc2, false, false);
    acc3 = __builtin_amdgcn_wmma_f32_16x16x32_bf16(false, a.bf, false, b3.bf, (short)0, acc3, false, false);

    // stash partial accumulators
#pragma unroll
    for (int r = 0; r < 8; ++r) {
        red[wave * 1024 + 0 * 256 + r * 32 + lane] = acc0[r];
        red[wave * 1024 + 1 * 256 + r * 32 + lane] = acc1[r];
        red[wave * 1024 + 2 * 256 + r * 32 + lane] = acc2[r];
        red[wave * 1024 + 3 * 256 + r * 32 + lane] = acc3[r];
    }
    __syncthreads();

    // cross-wave reduce + epilogue: 1024 elements, 8 per thread
#pragma unroll
    for (int q = 0; q < 8; ++q) {
        int e  = threadIdx.x * 8 + q;        // e = f*256 + r*32 + ln
        int f  = e >> 8;
        int r  = (e >> 5) & 7;
        int ln = e & 31;
        float s = red[0 * 1024 + e] + red[1 * 1024 + e] + red[2 * 1024 + e] + red[3 * 1024 + e];
        int j = j0 + r + 8 * (ln >> 4);      // C/D layout: VGPR r, lane half -> M = r + 8h
        int o = ln & 15;
        int c = f * 16 + o;
        // diagonal self-loop correction: + (A_fix[j,j]-A[j,j]) * dinv[j]*h[b,j,o]
        s += dfix[j] * bf16_to_f32(HsT[(size_t)c * NF + j]);
        float v = dinv[j] * s + b_gcn[o];
        Hrelu[((size_t)f * NF + j) * DOUT + o] = v > 0.f ? v : 0.f;
    }
}

// ---- Kernel D: out[b,g,:] = concat_f(Hrelu[b, g*10+f, :]) @ W_out + b_out ----
__global__ void out_proj(const float* __restrict__ Hrelu, const float* __restrict__ Wout,
                         const float* __restrict__ bout, float* __restrict__ out) {
    int t = blockIdx.x * blockDim.x + threadIdx.x;
    if (t >= BN * NN * DOUT) return;
    int od = t & 15;
    int g  = (t >> 4) & (NN - 1);
    int b  = t >> 14;
    float s = bout[od];
    const float* hr = Hrelu + ((size_t)b * NF + (size_t)g * FF) * DOUT;  // 160 contiguous
#pragma unroll
    for (int k = 0; k < FF * DOUT; ++k) s += hr[k] * Wout[k * DOUT + od];
    out[t] = s;
}

extern "C" void kernel_launch(void* const* d_in, const int* in_sizes, int n_in,
                              void* d_out, int out_size, void* d_ws, size_t ws_size,
                              hipStream_t stream) {
    (void)in_sizes; (void)n_in; (void)out_size; (void)ws_size;
    const float* x  = (const float*)d_in[0];
    const float* S  = (const float*)d_in[3];
    const float* Wg = (const float*)d_in[6];
    const float* bg = (const float*)d_in[7];
    const float* Wo = (const float*)d_in[8];
    const float* bo = (const float*)d_in[9];
    float* out = (float*)d_out;

    // workspace layout (~3.84 MB total)
    float*          dinv  = (float*)d_ws;                       // NF f32 (deg, then d^-1/2)
    float*          dfix  = dinv + NF;                          // NF f32
    unsigned short* HsT   = (unsigned short*)(dfix + NF);       // 64 x NF bf16
    float*          Hrelu = (float*)(HsT + (size_t)64 * NF);    // B x NF x 16 f32

    hipMemsetAsync(dinv, 0, NF * sizeof(float), stream);
    deg_partial <<<dim3(NF / 256, 32), 256, 0, stream>>>(S, dinv);
    finalize_deg<<<NF / 256, 256, 0, stream>>>(S, dinv, dfix);
    gcn_linear  <<<(64 * NF) / 256, 256, 0, stream>>>(x, Wg, dinv, HsT);
    gcn_agg_wmma<<<NF / 16, 128, 0, stream>>>(S, HsT, dinv, dfix, bg, Hrelu);
    out_proj    <<<(BN * NN * DOUT) / 256, 256, 0, stream>>>(Hrelu, Wo, bo, out);
}
